// GenericLinear_26938034881286
// MI455X (gfx1250) — matrix operand order repaired and checked by reading, not compiled, not run on hardware
//
#include <hip/hip_runtime.h>

// ---------------------------------------------------------------------------
// Fused geometric-algebra Dense layer for MI455X (gfx1250, wave32).
//
//   out[k*B + b, d] = bias[k*D+d] + sum_i S[i][k] * (x[i*B+b,:] @ W[:, J[i][k]*D+d])
//
// Pass 1: x (f32) -> bf16 in ws;  W (f32) -> transposed bf16 W^T in ws.
// Pass 2: WMMA GEMM, staging A/B tiles into LDS with the Tensor Data Mover
//         (tensor_load_to_lds + s_wait_tensorcnt), double-buffered; Cayley
//         sign applied as a bf16 sign-bit XOR on B fragments; bias in epilogue.
// ---------------------------------------------------------------------------

typedef __attribute__((ext_vector_type(16))) __bf16 v16bf;
typedef __attribute__((ext_vector_type(8)))  float  v8f;
typedef __attribute__((ext_vector_type(4)))  unsigned int v4u;
typedef __attribute__((ext_vector_type(8)))  int    v8i;
typedef __attribute__((ext_vector_type(4)))  int    v4i;

namespace ga {
constexpr int M8   = 8;
constexpr int Bb   = 2048;                 // batch per blade
constexpr int DIN  = 2048;
constexpr int DOUT = 2048;
constexpr int U    = M8 * DOUT;            // 16384
constexpr int BM = 128, BN = 128, BK = 64; // block tile
constexpr int LDA  = 72;                   // u16 pitch (144 B): 64 elems + 16 B pad
constexpr int ASZ  = 128 * LDA;            // u16 elements per LDS buffer
constexpr int NIT  = M8 * (DIN / BK);      // 256 staged tiles
}

__constant__ int c_J[8][8] = {
  {0,1,2,3,4,5,6,7},
  {1,0,4,5,2,3,7,6},
  {2,4,0,6,1,7,3,5},
  {3,5,6,0,7,1,2,4},
  {4,2,1,7,0,6,5,3},
  {5,3,7,1,6,0,4,2},
  {6,7,3,2,5,4,0,1},
  {7,6,5,4,3,2,1,0}};
#define SM 0x80008000u
__constant__ unsigned c_Sm[8][8] = {   // sign-bit mask for packed bf16 pairs
  { 0, 0, 0, 0, 0, 0, 0, 0},
  { 0, 0, 0, 0, 0, 0, 0, 0},
  { 0,SM, 0, 0,SM,SM, 0,SM},
  { 0,SM,SM, 0, 0,SM,SM, 0},
  {SM, 0,SM,SM, 0, 0,SM, 0},
  {SM, 0, 0,SM,SM, 0, 0,SM},
  {SM,SM, 0,SM, 0,SM, 0, 0},
  {SM,SM, 0,SM, 0,SM, 0, 0}};
#undef SM

__device__ __forceinline__ unsigned short f2bf(float f) {
  unsigned int u = __float_as_uint(f);
  u += 0x7FFFu + ((u >> 16) & 1u);         // round-to-nearest-even
  return (unsigned short)(u >> 16);
}

union FragU { uint4 q[2]; v16bf v; };

// ---- pass 1a: x f32 -> bf16 ------------------------------------------------
__global__ __launch_bounds__(256)
void convert_x_kernel(const float4* __restrict__ X4, uint2* __restrict__ O4, int n4) {
  for (int i = blockIdx.x * blockDim.x + threadIdx.x; i < n4; i += gridDim.x * blockDim.x) {
    const float4 f = X4[i];
    uint2 o;
    o.x = (unsigned)f2bf(f.x) | ((unsigned)f2bf(f.y) << 16);
    o.y = (unsigned)f2bf(f.z) | ((unsigned)f2bf(f.w) << 16);
    O4[i] = o;
  }
}

// ---- pass 1b: W [DIN,U] f32 -> W^T [U,DIN] bf16 (32x32 LDS tile transpose) -
__global__ __launch_bounds__(256)
void convert_wT_kernel(const float* __restrict__ W, unsigned short* __restrict__ WT) {
  using namespace ga;
  __shared__ float tile[32][33];
  const int tx = threadIdx.x, ty = threadIdx.y;       // block (32,8)
  const int c0 = blockIdx.x * 32;                     // W col  (U axis)
  const int r0 = blockIdx.y * 32;                     // W row  (DIN axis)
  #pragma unroll
  for (int j = 0; j < 4; ++j)
    tile[ty + j * 8][tx] = W[(size_t)(r0 + ty + j * 8) * U + c0 + tx];
  __syncthreads();
  #pragma unroll
  for (int j = 0; j < 4; ++j)
    WT[(size_t)(c0 + ty + j * 8) * DIN + r0 + tx] = f2bf(tile[tx][ty + j * 8]);
}

// ---- TDM issue: one 64x128-elem bf16 tile -> LDS (padded to 144B pitch) ----
__device__ __forceinline__ void tdm_issue(unsigned long long gaddr, unsigned ldsaddr,
                                          v8i g1) {
  v4u g0;
  g0.x = 1u;                                             // count=1, user mode
  g0.y = ldsaddr;                                        // LDS byte address
  g0.z = (unsigned)gaddr;                                // global_addr[31:0]
  g0.w = ((unsigned)(gaddr >> 32) & 0x01FFFFFFu) | 0x80000000u;  // [56:32] | type=2
  v4i z4 = {0, 0, 0, 0};
  v8i z8 = {0, 0, 0, 0, 0, 0, 0, 0};
  __builtin_amdgcn_tensor_load_to_lds(g0, g1, z4, z4, z8, 0);   // 6-arg (clang-23)
}

// ---- pass 2: fused WMMA GEMM ----------------------------------------------
__global__ __launch_bounds__(256)
void ga_dense_wmma_kernel(const unsigned short* __restrict__ Xbf,  // [8B, DIN] bf16
                          const unsigned short* __restrict__ WT,   // [U, DIN]  bf16
                          const float* __restrict__ bias,
                          float*       __restrict__ out) {
  using namespace ga;
  __shared__ unsigned short As[2 * ASZ];   // [buf][row 0..127][k 0..63] pitch 72
  __shared__ unsigned short Bs[2 * ASZ];   // [buf][n   0..127][k 0..63] pitch 72

  const int tid  = threadIdx.x;
  const int lane = tid & 31;
  const int wv   = tid >> 5;
  const int wrow = wv >> 2;                 // 2 x 64 rows
  const int wcol = wv & 3;                  // 4 x 32 cols

  const int dtile = blockIdx.x * BN;
  const int kb    = blockIdx.y;
  const int k     = kb >> 4;                // output blade
  const int brow0 = (kb & 15) * BM;

  // Shared D# group1: data_size=2B, pad every 32 DWORDs (one 128B row) by 4 DWORDs,
  // tensor = [16384 rows x 2048 elems], tile = [128 rows x 64 elems], stride 2048.
  v8i g1;
  g1[0] = (1 << 16) | (1 << 20) | (4 << 22) | (3 << 25);
  g1[1] = (int)((unsigned)DIN << 16);       // tensor_dim0 = 2048 (low16 @ [63:48])
  g1[2] = (int)((unsigned)U << 16);         // tensor_dim1 = 16384 (low16 @ [95:80])
  g1[3] = (BK << 16);                       // tile_dim0 = 64
  g1[4] = BM;                               // tile_dim1 = 128, tile_dim2 = 0
  g1[5] = DIN;                              // tensor_dim0_stride = 2048
  g1[6] = 0;
  g1[7] = 0;

  const unsigned ldsA0 = (unsigned)(unsigned long long)(void*)As;
  const unsigned ldsB0 = (unsigned)(unsigned long long)(void*)Bs;
  const unsigned long long xb = (unsigned long long)(const void*)Xbf;
  const unsigned long long wb = (unsigned long long)(const void*)WT;

  v8f acc[4][2];
  #pragma unroll
  for (int mr = 0; mr < 4; ++mr)
    #pragma unroll
    for (int nc = 0; nc < 2; ++nc)
      #pragma unroll
      for (int e = 0; e < 8; ++e) acc[mr][nc][e] = 0.0f;

  // prologue: stage tile 0 into buffer 0
  if (tid == 0) {
    const int jn = c_J[0][k];
    tdm_issue(xb + ((size_t)brow0 * DIN) * 2,                 ldsA0, g1);
    tdm_issue(wb + ((size_t)(jn * DOUT + dtile) * DIN) * 2,   ldsB0, g1);
  }

  #pragma unroll 1
  for (int t = 0; t < NIT; ++t) {
    if (tid == 0) {
      if (t + 1 < NIT) {
        const int tn  = t + 1;
        const int ii  = tn >> 5;
        const int kk2 = (tn & 31) << 6;
        const int jn  = c_J[ii][k];
        const unsigned boff = (unsigned)((tn & 1) * ASZ * 2);
        tdm_issue(xb + ((size_t)(ii * Bb + brow0) * DIN + kk2) * 2,       ldsA0 + boff, g1);
        tdm_issue(wb + ((size_t)(jn * DOUT + dtile) * DIN + kk2) * 2,     ldsB0 + boff, g1);
        __builtin_amdgcn_s_wait_tensorcnt(2);   // tile t's 2 DMAs complete
      } else {
        __builtin_amdgcn_s_wait_tensorcnt(0);
      }
    }
    __syncthreads();                            // tile t visible to all waves

    const unsigned short* as = As + (t & 1) * ASZ;
    const unsigned short* bs = Bs + (t & 1) * ASZ;
    const unsigned smask = c_Sm[t >> 5][k];     // Cayley sign for blade i = t>>5

    #pragma unroll
    for (int half = 0; half < 2; ++half) {      // two K=32 slabs per staged tile
      FragU a[4];
      #pragma unroll
      for (int mr = 0; mr < 4; ++mr) {
        const int row  = wrow * 64 + mr * 16 + (lane & 15);
        const int base = row * LDA + half * 32 + ((lane & 16) ? 8 : 0);
        a[mr].q[0] = *(const uint4*)&as[base];        // K {0..7} / {8..15}
        a[mr].q[1] = *(const uint4*)&as[base + 16];   // K {16..23} / {24..31}
      }
      FragU b[2];
      #pragma unroll
      for (int nc = 0; nc < 2; ++nc) {
        const int n    = wcol * 32 + nc * 16 + (lane & 15);
        const int base = n * LDA + half * 32 + ((lane & 16) ? 8 : 0);
        b[nc].q[0] = *(const uint4*)&bs[base];
        b[nc].q[1] = *(const uint4*)&bs[base + 16];
        b[nc].q[0].x ^= smask; b[nc].q[0].y ^= smask;   // fold sign: s*(A@B)=A@(s*B)
        b[nc].q[0].z ^= smask; b[nc].q[0].w ^= smask;
        b[nc].q[1].x ^= smask; b[nc].q[1].y ^= smask;
        b[nc].q[1].z ^= smask; b[nc].q[1].w ^= smask;
      }
      #pragma unroll
      for (int mr = 0; mr < 4; ++mr)
        #pragma unroll
        for (int nc = 0; nc < 2; ++nc)
          acc[mr][nc] = __builtin_amdgcn_wmma_f32_16x16x32_bf16(
              false, a[mr].v, false, b[nc].v,
              (short)0, acc[mr][nc], false, false);
    }
    __syncthreads();                            // reads done before buffer reuse
  }

  // epilogue: + bias (blade-0 Cayley row is identity => just bias[k*D+d])
  const float* bptr = bias + (size_t)k * DOUT + dtile;
  #pragma unroll
  for (int nc = 0; nc < 2; ++nc) {
    const int n_l = wcol * 32 + nc * 16 + (lane & 15);
    const float bv = bptr[n_l];
    #pragma unroll
    for (int mr = 0; mr < 4; ++mr) {
      #pragma unroll
      for (int v = 0; v < 8; ++v) {
        const int m_l = wrow * 64 + mr * 16 + v + ((lane & 16) ? 8 : 0);
        const size_t row = (size_t)(k * Bb + brow0 + m_l);
        out[row * DOUT + dtile + n_l] = acc[mr][nc][v] + bv;
      }
    }
  }
}

extern "C" void kernel_launch(void* const* d_in, const int* in_sizes, int n_in,
                              void* d_out, int out_size, void* d_ws, size_t ws_size,
                              hipStream_t stream) {
  using namespace ga;
  const float* X    = (const float*)d_in[0];   // [16384, 2048] f32
  const float* Wm   = (const float*)d_in[1];   // [2048, 16384] f32
  const float* bias = (const float*)d_in[2];   // [16384] f32
  float*       out  = (float*)d_out;           // [16384, 2048] f32

  unsigned short* xbf = (unsigned short*)d_ws;                    // 67 MB
  unsigned short* wt  = xbf + (size_t)M8 * Bb * DIN;              // 67 MB

  // pass 1a: x -> bf16
  const int n4 = (M8 * Bb * DIN) / 4;
  convert_x_kernel<<<4096, 256, 0, stream>>>((const float4*)X, (uint2*)xbf, n4);
  // pass 1b: W -> W^T bf16
  convert_wT_kernel<<<dim3(U / 32, DIN / 32), dim3(32, 8), 0, stream>>>(Wm, wt);
  // pass 2: fused TDM + WMMA GEMM
  dim3 grid(DOUT / BN, (M8 * Bb) / BM);        // (16, 128)
  ga_dense_wmma_kernel<<<grid, 256, 0, stream>>>(xbf, wt, bias, out);
}
